// LiberviaAttention_36910948942418
// MI455X (gfx1250) — compile-verified, matrix operand order checked by reading, not run on hardware
//
#include <hip/hip_runtime.h>
#include <hip/hip_bf16.h>

// ---------------------------------------------------------------------------
// MI455X (gfx1250) fused attention layer, round 2.
// Compute-bound (~60 GFLOP vs ~60MB traffic) => bf16 WMMA everywhere.
// New: one-time bf16 conversion / weight transposition so all GEMM + flash
// tiles are staged with GLOBAL_LOAD_ASYNC_TO_LDS_B128 (ASYNCcnt) and
// double-buffered: async fetch of tile k+1 overlaps WMMA on tile k.
// ---------------------------------------------------------------------------

typedef __attribute__((ext_vector_type(8)))  __bf16 v8bf;
typedef __attribute__((ext_vector_type(16))) __bf16 v16bf;
typedef __attribute__((ext_vector_type(8)))  float  v8f;

#define S_LEN 2048
#define D_MODEL 2048
#define NHEAD 32
#define NKVH 8
#define HD 64

__device__ __forceinline__ v8f zero8() {
  v8f z;
#pragma unroll
  for (int i = 0; i < 8; ++i) z[i] = 0.0f;
  return z;
}

__device__ __forceinline__ v8f wmma_bf16(v16bf a, v16bf b, v8f c) {
  return __builtin_amdgcn_wmma_f32_16x16x32_bf16(
      false, a, false, b, (short)0, c, false, false);
}

// Low 32 bits of a generic shared-space address are the LDS byte address.
#define LDS_ADDR(p) ((unsigned)(unsigned long long)(p))

// Async global->LDS 16B copy (GVS mode: 64-bit SGPR base + 32-bit VGPR off).
__device__ __forceinline__ void async_b128(unsigned lds, unsigned long long gbase,
                                           unsigned voff) {
  asm volatile("global_load_async_to_lds_b128 %0, %1, %2"
               :: "v"(lds), "v"(voff), "s"(gbase) : "memory");
}
__device__ __forceinline__ void wait_async0() {
  asm volatile("s_wait_asynccnt 0" ::: "memory");
}

// Load one 16x32 bf16 fragment from an LDS row (row-major, K contiguous).
// Lanes 0-15: K=0..7 / 16..23; lanes 16-31: K=8..15 / 24..31 -> 2x ds_load_b128.
__device__ __forceinline__ v16bf ld_frag16(const __bf16* rowptr, int lane) {
  const int klo = (lane >> 4) << 3;
  v8bf lo = *reinterpret_cast<const v8bf*>(rowptr + klo);
  v8bf hi = *reinterpret_cast<const v8bf*>(rowptr + 16 + klo);
  v16bf r;
#pragma unroll
  for (int i = 0; i < 8; ++i) { r[i] = lo[i]; r[i + 8] = hi[i]; }
  return r;
}

// ---------------------------------------------------------------------------
// One-time converters (bandwidth noise: ~40MB total at 23.3 TB/s).
// ---------------------------------------------------------------------------
__global__ void cvt_bf16_kernel(const float* __restrict__ in,
                                __bf16* __restrict__ out, int n) {
  int idx = blockIdx.x * 256 + threadIdx.x;
  if (idx < n) out[idx] = (__bf16)in[idx];
}

// out[n][k] = in[k][n]; in fp32 [Kd][Nd], out bf16 [Nd][Kd]. Output-coalesced.
__global__ void transpose_cvt_kernel(const float* __restrict__ in,
                                     __bf16* __restrict__ out, int Kd, int Nd) {
  int idx = blockIdx.x * 256 + threadIdx.x;
  if (idx >= Kd * Nd) return;
  int n = idx / Kd;
  int k = idx % Kd;
  out[idx] = (__bf16)in[(size_t)k * Nd + n];
}

// ---------------------------------------------------------------------------
// GEMM: C[M,N] = A[M,2048] * Bt[N,2048]^T, bf16 in, WMMA fp32 accumulate.
// 128x128x32 block, 8 waves (2x4), wave tile 64x32. Tiles staged with
// async-to-LDS, double buffered. OMODE: 0 = fp32 C, 1 = bf16 C,
// 2 = bf16 C transposed ([N][2048], used to produce V^T).
// ---------------------------------------------------------------------------
template <int OMODE, int N_>
__global__ __launch_bounds__(256) void gemm_async_wmma(
    const __bf16* __restrict__ A, const __bf16* __restrict__ Bt,
    void* __restrict__ Cout) {
  constexpr int K_ = 2048;
  __shared__ __align__(16) __bf16 As[2][128][32];
  __shared__ __align__(16) __bf16 Bs[2][128][32];

  const int tid  = threadIdx.x;
  const int lane = tid & 31;
  const int wave = tid >> 5;
  const int wm   = wave >> 2;
  const int wn   = wave & 3;
  const int m0   = blockIdx.y * 128;
  const int n0   = blockIdx.x * 128;
  const int half = lane >> 4;
  const int nl   = lane & 15;

  const unsigned long long aB = (unsigned long long)A;
  const unsigned long long bB = (unsigned long long)Bt;

  auto issue = [&](int buf, int k0) {
    const unsigned al = LDS_ADDR(&As[buf][0][0]);
    const unsigned bl = LDS_ADDR(&Bs[buf][0][0]);
#pragma unroll
    for (int i = 0; i < 2; ++i) {
      int ch = tid + i * 256;          // 512 chunks of 16B per 8KB tile
      int r  = ch >> 2;                // row (64B rows)
      int c  = (ch & 3) << 4;          // byte offset in row
      async_b128(al + r * 64 + c, aB, (unsigned)((m0 + r) * (K_ * 2) + k0 * 2 + c));
      async_b128(bl + r * 64 + c, bB, (unsigned)((n0 + r) * (K_ * 2) + k0 * 2 + c));
    }
  };

  v8f acc[4][2];
#pragma unroll
  for (int mt = 0; mt < 4; ++mt)
#pragma unroll
    for (int nt = 0; nt < 2; ++nt) acc[mt][nt] = zero8();

  issue(0, 0);
  wait_async0();
  __syncthreads();

  for (int kb = 0; kb < K_ / 32; ++kb) {
    const int cur = kb & 1;
    if (kb + 1 < K_ / 32) issue(cur ^ 1, (kb + 1) * 32);  // overlap with WMMA

    v16bf af[4], bfg[2];
#pragma unroll
    for (int mt = 0; mt < 4; ++mt)
      af[mt] = ld_frag16(&As[cur][wm * 64 + mt * 16 + nl][0], lane);
#pragma unroll
    for (int nt = 0; nt < 2; ++nt)
      bfg[nt] = ld_frag16(&Bs[cur][wn * 32 + nt * 16 + nl][0], lane);
#pragma unroll
    for (int mt = 0; mt < 4; ++mt)
#pragma unroll
      for (int nt = 0; nt < 2; ++nt)
        acc[mt][nt] = wmma_bf16(af[mt], bfg[nt], acc[mt][nt]);

    if (kb + 1 < K_ / 32) wait_async0();
    __syncthreads();
  }

  const int row0 = m0 + wm * 64 + 8 * half;
  const int col0 = n0 + wn * 32 + nl;
#pragma unroll
  for (int mt = 0; mt < 4; ++mt)
#pragma unroll
    for (int nt = 0; nt < 2; ++nt)
#pragma unroll
      for (int r = 0; r < 8; ++r) {
        int row = row0 + mt * 16 + r;
        int col = col0 + nt * 16;
        float v = acc[mt][nt][r];
        if constexpr (OMODE == 0) {
          ((float*)Cout)[(size_t)row * N_ + col] = v;
        } else if constexpr (OMODE == 1) {
          ((__bf16*)Cout)[(size_t)row * N_ + col] = (__bf16)v;
        } else {
          ((__bf16*)Cout)[(size_t)col * 2048 + row] = (__bf16)v;  // V^T
        }
      }
}

// ---------------------------------------------------------------------------
// RoPE in place on bf16 [s][head*64+d].
// ---------------------------------------------------------------------------
__global__ void rope_kernel(__bf16* __restrict__ x, int nh) {
  int idx = blockIdx.x * 256 + threadIdx.x;
  int total = S_LEN * nh * 32;
  if (idx >= total) return;
  int i = idx & 31;
  int t = idx >> 5;
  int h = t % nh;
  int s = t / nh;
  float inv = __expf(-(float)i * (9.210340371976184f / 32.0f));
  float ang = (float)s * inv;
  float sn, cs;
  __sincosf(ang, &sn, &cs);
  __bf16* p = x + (size_t)s * nh * HD + (size_t)h * HD;
  float x1 = (float)p[i], x2 = (float)p[i + 32];
  p[i]      = (__bf16)(x1 * cs - x2 * sn);
  p[i + 32] = (__bf16)(x2 * cs + x1 * sn);
}

// ---------------------------------------------------------------------------
// Flash attention, causal, GQA. Grid (S/128, NHEAD), 8 waves own 16 q-rows
// each. Q staged async once; K / V^T tiles async double-buffered. Q's LDS is
// dead after per-wave fragment extraction -> aliased as the P buffer
// (wave-private rows, same-wave DS ordering).
// ---------------------------------------------------------------------------
__global__ __launch_bounds__(256) void flash_attn_kernel(
    const __bf16* __restrict__ Q, const __bf16* __restrict__ Kg,
    const __bf16* __restrict__ Vt, __bf16* __restrict__ O) {
  __shared__ __align__(16) __bf16 QP[128][64];    // Q tile, then P tile
  __shared__ __align__(16) __bf16 Ks[2][64][64];  // [kv][d]
  __shared__ __align__(16) __bf16 Vs[2][64][64];  // [d][kv]

  const int tid  = threadIdx.x;
  const int lane = tid & 31;
  const int wave = tid >> 5;
  const int qb   = blockIdx.x;
  const int h    = blockIdx.y;
  const int kvh  = h >> 2;
  const int q0   = qb * 128;
  const int half = lane >> 4;
  const int nl   = lane & 15;

  const unsigned long long qB = (unsigned long long)Q;
  const unsigned long long kB = (unsigned long long)Kg;
  const unsigned long long vB = (unsigned long long)Vt;

  // Q tile: 128 rows x 128B (row stride in q = 2048*2 bytes).
  {
    const unsigned ql = LDS_ADDR(&QP[0][0]);
#pragma unroll
    for (int i = 0; i < 4; ++i) {
      int ch = tid + i * 256;   // 1024 chunks
      int r  = ch >> 3;
      int c  = (ch & 7) << 4;
      async_b128(ql + r * 128 + c,
                 qB, (unsigned)((q0 + r) * (D_MODEL * 2) + h * 128 + c));
    }
  }
  auto issueKV = [&](int buf, int kb) {
    const unsigned kl = LDS_ADDR(&Ks[buf][0][0]);
    const unsigned vl = LDS_ADDR(&Vs[buf][0][0]);
#pragma unroll
    for (int i = 0; i < 2; ++i) {
      int ch = tid + i * 256;   // 512 chunks per 8KB tile
      int r  = ch >> 3;
      int c  = (ch & 7) << 4;
      // K: [s][kvh*64+d], row stride 512*2 B
      async_b128(kl + r * 128 + c,
                 kB, (unsigned)((kb * 64 + r) * (NKVH * HD * 2) + kvh * 128 + c));
      // V^T: [d][s], row stride 2048*2 B
      async_b128(vl + r * 128 + c,
                 vB, (unsigned)((kvh * 64 + r) * (S_LEN * 2) + kb * 128 + c));
    }
  };

  issueKV(0, 0);
  wait_async0();
  __syncthreads();

  v16bf qf[2];
#pragma unroll
  for (int kk = 0; kk < 2; ++kk)
    qf[kk] = ld_frag16(&QP[wave * 16 + nl][kk * 32], lane);

  v8f ot[4];
#pragma unroll
  for (int j = 0; j < 4; ++j) ot[j] = zero8();
  float mrow[8], lrow[8];
#pragma unroll
  for (int r = 0; r < 8; ++r) { mrow[r] = -3.0e38f; lrow[r] = 0.0f; }

  const int nkb = 2 * (qb + 1);
  for (int kb = 0; kb < nkb; ++kb) {
    const int cur = kb & 1;
    if (kb + 1 < nkb) issueKV(cur ^ 1, kb + 1);

    // S = Q K^T (16x64 per wave)
    v8f st[4];
#pragma unroll
    for (int j = 0; j < 4; ++j) {
      v16bf kf0 = ld_frag16(&Ks[cur][j * 16 + nl][0], lane);
      v16bf kf1 = ld_frag16(&Ks[cur][j * 16 + nl][32], lane);
      v8f s = zero8();
      s = wmma_bf16(qf[0], kf0, s);
      s = wmma_bf16(qf[1], kf1, s);
      st[j] = s;
    }

    // Online softmax (row stats across 16-lane halves).
#pragma unroll
    for (int r = 0; r < 8; ++r) {
      const int qg = q0 + wave * 16 + r + 8 * half;
      float rm = -3.0e38f;
#pragma unroll
      for (int j = 0; j < 4; ++j) {
        float sv = st[j][r] * 0.125f;
        int kg = kb * 64 + j * 16 + nl;
        if (kg > qg) sv = -3.0e38f;
        st[j][r] = sv;
        rm = fmaxf(rm, sv);
      }
#pragma unroll
      for (int m = 1; m < 16; m <<= 1) rm = fmaxf(rm, __shfl_xor(rm, m, 32));
      float mnew  = fmaxf(mrow[r], rm);
      float alpha = __expf(mrow[r] - mnew);
      mrow[r] = mnew;
      float rs = 0.0f;
#pragma unroll
      for (int j = 0; j < 4; ++j) {
        float p = __expf(st[j][r] - mnew);
        st[j][r] = p;
        rs += p;
      }
#pragma unroll
      for (int m = 1; m < 16; m <<= 1) rs += __shfl_xor(rs, m, 32);
      lrow[r] = lrow[r] * alpha + rs;
#pragma unroll
      for (int j = 0; j < 4; ++j) ot[j][r] *= alpha;
    }

    // P -> LDS (wave-private rows of QP), reload as A fragment.
#pragma unroll
    for (int r = 0; r < 8; ++r)
#pragma unroll
      for (int j = 0; j < 4; ++j)
        QP[wave * 16 + r + 8 * half][j * 16 + nl] = (__bf16)st[j][r];
    asm volatile("s_wait_dscnt 0" ::: "memory");

    v16bf pf[2];
#pragma unroll
    for (int kk = 0; kk < 2; ++kk)
      pf[kk] = ld_frag16(&QP[wave * 16 + nl][kk * 32], lane);
#pragma unroll
    for (int j = 0; j < 4; ++j) {
      v16bf vf0 = ld_frag16(&Vs[cur][j * 16 + nl][0], lane);
      v16bf vf1 = ld_frag16(&Vs[cur][j * 16 + nl][32], lane);
      ot[j] = wmma_bf16(pf[0], vf0, ot[j]);
      ot[j] = wmma_bf16(pf[1], vf1, ot[j]);
    }

    if (kb + 1 < nkb) wait_async0();
    __syncthreads();
  }

  // Normalize, write context bf16 [s][h*64+d].
#pragma unroll
  for (int r = 0; r < 8; ++r) {
    float invl = 1.0f / lrow[r];
    int row = q0 + wave * 16 + r + 8 * half;
#pragma unroll
    for (int j = 0; j < 4; ++j)
      O[(size_t)row * D_MODEL + h * HD + j * 16 + nl] =
          (__bf16)(ot[j][r] * invl);
  }
}

// ---------------------------------------------------------------------------
// Launch. Workspace (bf16): hs 8MB | WqT 8MB | WkT 2MB | WvT 2MB | WoT 8MB |
// q 8MB | k 2MB | vT 2MB | ctx 8MB = 48MB.
// ---------------------------------------------------------------------------
extern "C" void kernel_launch(void* const* d_in, const int* in_sizes, int n_in,
                              void* d_out, int out_size, void* d_ws,
                              size_t ws_size, hipStream_t stream) {
  const float* hs = (const float*)d_in[0];
  const float* Wq = (const float*)d_in[1];
  const float* Wk = (const float*)d_in[2];
  const float* Wv = (const float*)d_in[3];
  const float* Wo = (const float*)d_in[4];
  float* out = (float*)d_out;

  char* ws = (char*)d_ws;
  size_t off = 0;
  __bf16* hsB = (__bf16*)(ws + off); off += (size_t)S_LEN * D_MODEL * 2;
  __bf16* WqT = (__bf16*)(ws + off); off += (size_t)D_MODEL * D_MODEL * 2;
  __bf16* WkT = (__bf16*)(ws + off); off += (size_t)(NKVH * HD) * D_MODEL * 2;
  __bf16* WvT = (__bf16*)(ws + off); off += (size_t)(NKVH * HD) * D_MODEL * 2;
  __bf16* WoT = (__bf16*)(ws + off); off += (size_t)D_MODEL * D_MODEL * 2;
  __bf16* q   = (__bf16*)(ws + off); off += (size_t)S_LEN * D_MODEL * 2;
  __bf16* k   = (__bf16*)(ws + off); off += (size_t)S_LEN * NKVH * HD * 2;
  __bf16* vT  = (__bf16*)(ws + off); off += (size_t)(NKVH * HD) * S_LEN * 2;
  __bf16* ctx = (__bf16*)(ws + off);

  dim3 blk(256);
  // One-time precision/layout conversion.
  cvt_bf16_kernel<<<(S_LEN * D_MODEL + 255) / 256, blk, 0, stream>>>(
      hs, hsB, S_LEN * D_MODEL);
  transpose_cvt_kernel<<<(D_MODEL * D_MODEL + 255) / 256, blk, 0, stream>>>(
      Wq, WqT, D_MODEL, D_MODEL);
  transpose_cvt_kernel<<<(D_MODEL * NKVH * HD + 255) / 256, blk, 0, stream>>>(
      Wk, WkT, D_MODEL, NKVH * HD);
  transpose_cvt_kernel<<<(D_MODEL * NKVH * HD + 255) / 256, blk, 0, stream>>>(
      Wv, WvT, D_MODEL, NKVH * HD);
  transpose_cvt_kernel<<<(D_MODEL * D_MODEL + 255) / 256, blk, 0, stream>>>(
      Wo, WoT, D_MODEL, D_MODEL);
  // Projections (V written transposed for the flash V^T path).
  gemm_async_wmma<1, D_MODEL><<<dim3(D_MODEL / 128, S_LEN / 128), blk, 0, stream>>>(
      hsB, WqT, q);
  gemm_async_wmma<1, NKVH * HD><<<dim3(NKVH * HD / 128, S_LEN / 128), blk, 0, stream>>>(
      hsB, WkT, k);
  gemm_async_wmma<2, NKVH * HD><<<dim3(NKVH * HD / 128, S_LEN / 128), blk, 0, stream>>>(
      hsB, WvT, vT);
  // RoPE
  rope_kernel<<<(S_LEN * NHEAD * 32 + 255) / 256, blk, 0, stream>>>(q, NHEAD);
  rope_kernel<<<(S_LEN * NKVH * 32 + 255) / 256, blk, 0, stream>>>(k, NKVH);
  // Attention
  flash_attn_kernel<<<dim3(S_LEN / 128, NHEAD), blk, 0, stream>>>(q, k, vT, ctx);
  // Output projection (fp32 out).
  gemm_async_wmma<0, D_MODEL><<<dim3(D_MODEL / 128, S_LEN / 128), blk, 0, stream>>>(
      ctx, WoT, out);
}
